// YOLODetectionHead_44770739093585
// MI455X (gfx1250) — matrix operand order repaired and checked by reading, not compiled, not run on hardware
//
#include <hip/hip_runtime.h>
#include <hip/hip_bf16.h>
#include <math.h>

typedef __attribute__((ext_vector_type(16))) __bf16 v16bf;
typedef __attribute__((ext_vector_type(8)))  float  v8f;

#define NPRED   25200
#define KPRE    2048
#define MAXDET  300
#define CONF_T  1e-4f
#define NMS_IOU 0.45f
#define CLS_OFF 4096.0f
#define NSORT   32768

#if defined(__gfx1250__) && \
    __has_builtin(__builtin_amdgcn_global_load_async_to_lds_b128) && \
    __has_builtin(__builtin_amdgcn_s_wait_asynccnt)
#define USE_ASYNC_LDS 1
#else
#define USE_ASYNC_LDS 0
#endif

#if USE_ASYNC_LDS
// Builtin expects (int4 global*, int4 shared*, imm offset, imm cpol).
typedef int v4i_ __attribute__((vector_size(16)));
typedef __attribute__((address_space(1))) v4i_ g_v4i;
typedef __attribute__((address_space(3))) v4i_ l_v4i;
// generic global addr == AS1 addr numerically
__device__ __forceinline__ g_v4i* gbl_v4(const void* p) {
    return (g_v4i*)(unsigned long long)p;
}
// generic shared addr: low 32 bits are the LDS offset (ISA aperture rule)
__device__ __forceinline__ l_v4i* lds_v4(const void* p) {
    return (l_v4i*)(unsigned int)(unsigned long long)p;
}
#endif

__device__ __forceinline__ float sigmoidf_(float x) {
    return 1.0f / (1.0f + __expf(-x));
}

// ---------------------------------------------------------------------------
// Pack f32 weights (C x 255, row-major) into bf16 WMMA-B tiles.
// Layout: [kt][nt][lane][e] where this code's B k order (contiguous, same as A) is:
//   N = nt*16 + (lane & 15),  K = kt*32 + (lane>>4)*16 + e
// N is padded to 256 with zeros (tile 15 cols 240..255, col 255 = 0).
// ---------------------------------------------------------------------------
__global__ void yolo_pack_w(const float* __restrict__ w,
                            __bf16* __restrict__ out, int ktiles) {
    int p = blockIdx.x * blockDim.x + threadIdx.x;
    int total = ktiles * 8192;
    if (p >= total) return;
    int e    = p & 15;
    int lane = (p >> 4) & 31;
    int nt   = (p >> 9) & 15;
    int kt   = p >> 13;
    int n = nt * 16 + (lane & 15);
    int c = kt * 32 + ((lane >> 4) << 4) + e;
    float v = (n < 255) ? w[c * 255 + n] : 0.0f;
    out[p] = (__bf16)v;
}

// ---------------------------------------------------------------------------
// Fused 1x1-conv (bf16 WMMA, f32 accumulate) + YOLO decode.
// 128 threads = 4 waves; wave w owns M-tile [tile*64 + w*16, +16) of spatial
// positions, computes all 16 N-tiles (N=255 padded to 256), spills to LDS,
// then the block decodes 64 positions x 3 anchors.
// ---------------------------------------------------------------------------
__global__ void __launch_bounds__(128)
yolo_decode(const float* __restrict__ feat, const __bf16* __restrict__ wp,
            const float* __restrict__ bias, float* __restrict__ boxes,
            float* __restrict__ scores, int* __restrict__ labels,
            int C, int HW, int W, int nTiles, float stride, int scaleOff,
            float ax0, float ay0, float ax1, float ay1, float ax2, float ay2)
{
    extern __shared__ float lds[];                // 64 x 256 f32 = 64 KB
    const int tid  = threadIdx.x;
    const int wave = tid >> 5;
    const int lane = tid & 31;
    const int b    = blockIdx.x / nTiles;
    const int tile = blockIdx.x % nTiles;
    const int mBase = tile * 64;
    const int m0    = mBase + wave * 16;
    const float* fB = feat + (size_t)b * C * HW;
    const int ktiles = C >> 5;

    if (m0 + 16 <= HW) {                          // wave-uniform guard
        const int row  = lane & 15;               // A: M = row
        const int half = lane >> 4;               // A: K-half select
        const int m    = m0 + row;
        v8f acc[16] = {};
        for (int kt = 0; kt < ktiles; ++kt) {
            if (kt + 1 < ktiles)
                __builtin_prefetch(fB + (size_t)((kt + 1) * 32 + half * 8) * HW + m, 0, 0);
            // Gather A tile: contiguous k order used here (same as packed B)
            //   K = half*16 + e
            v16bf a;
#pragma unroll
            for (int e = 0; e < 16; ++e) {
                int kl = half * 16 + e;
                a[e] = (__bf16)fB[(size_t)(kt * 32 + kl) * HW + m];
            }
            const v16bf* bptr = (const v16bf*)(wp + (size_t)kt * 8192);
#pragma unroll
            for (int nt = 0; nt < 16; ++nt) {
                v16bf bm = bptr[nt * 32 + lane];
                acc[nt] = __builtin_amdgcn_wmma_f32_16x16x32_bf16(
                    false, a, false, bm, (short)0, acc[nt], false, false);
            }
        }
        // C/D layout: lane holds col N=lane&15; VGPR r holds M = r + 8*(lane>>4)
        const int col = lane & 15;
#pragma unroll
        for (int nt = 0; nt < 16; ++nt)
#pragma unroll
            for (int r = 0; r < 8; ++r)
                lds[(wave * 16 + r + 8 * half) * 256 + nt * 16 + col] = acc[nt][r];
    }
    __syncthreads();

    const float anchX[3] = {ax0, ax1, ax2};
    const float anchY[3] = {ay0, ay1, ay2};
    for (int pa = tid; pa < 192; pa += 128) {
        int a  = pa >> 6;                         // anchor 0..2
        int ml = pa & 63;
        int m  = mBase + ml;
        if (m >= HW) continue;
        const float* pr = &lds[ml * 256 + a * 85];
        const float* bi = bias + a * 85;
        float px  = sigmoidf_(pr[0] + bi[0]);
        float py  = sigmoidf_(pr[1] + bi[1]);
        float pw  = sigmoidf_(pr[2] + bi[2]);
        float ph  = sigmoidf_(pr[3] + bi[3]);
        float obj = sigmoidf_(pr[4] + bi[4]);
        float gx = (float)(m % W), gy = (float)(m / W);
        float cx = (px * 2.0f - 0.5f + gx) * stride;
        float cy = (py * 2.0f - 0.5f + gy) * stride;
        float ww = pw * 2.0f; ww = ww * ww * anchX[a];
        float hh = ph * 2.0f; hh = hh * hh * anchY[a];
        float best = -1.0f; int arg = 0;
        for (int c = 0; c < 80; ++c) {
            float s = obj * sigmoidf_(pr[5 + c] + bi[5 + c]);
            if (s > best) { best = s; arg = c; }
        }
        int gi = scaleOff + a * HW + m;
        size_t o4 = ((size_t)b * NPRED + gi) * 4;
        boxes[o4 + 0] = cx - ww * 0.5f;
        boxes[o4 + 1] = cy - hh * 0.5f;
        boxes[o4 + 2] = cx + ww * 0.5f;
        boxes[o4 + 3] = cy + hh * 0.5f;
        scores[(size_t)b * NPRED + gi] = best;
        labels[(size_t)b * NPRED + gi] = arg;
    }
}

// ---------------------------------------------------------------------------
// Per-image top-KPRE via full bitonic sort of 32768 (score,idx) pairs in LDS.
// 1024 threads, 256 KB dynamic LDS (CDNA5: 320 KB per WGP).
// Score staging uses GLOBAL_LOAD_ASYNC_TO_LDS_B128 (ASYNCcnt) when available.
// ---------------------------------------------------------------------------
__global__ void __launch_bounds__(1024)
yolo_topk(const float* __restrict__ scores, const int* __restrict__ labels,
          const float* __restrict__ boxes,
          float* __restrict__ cBoxes, float* __restrict__ cScores,
          int* __restrict__ cLabels)
{
    extern __shared__ char smem[];
    float* sS = (float*)smem;
    int*   sI = (int*)(smem + NSORT * 4);
    const int b = blockIdx.x, tid = threadIdx.x;
    const float* sc = scores + (size_t)b * NPRED;

#if USE_ASYNC_LDS
    // Async-DMA raw scores into LDS (NPRED % 4 == 0, 16B-aligned both sides).
    for (int i4 = tid; i4 < NPRED / 4; i4 += 1024)
        __builtin_amdgcn_global_load_async_to_lds_b128(
            gbl_v4(sc + i4 * 4), lds_v4(sS + i4 * 4), 0, 0);
    for (int i = NPRED + tid; i < NSORT; i += 1024) sS[i] = 0.0f;
    __builtin_amdgcn_s_wait_asynccnt(0);
    __syncthreads();
    // In-LDS threshold fixup + index init.
    for (int i = tid; i < NSORT; i += 1024) {
        float v = (i < NPRED) ? sS[i] : 0.0f;
        sS[i] = (v > CONF_T) ? v : 0.0f;
        sI[i] = (i < NPRED) ? i : 0;
    }
#else
    for (int i = tid; i < NSORT; i += 1024) {
        float v = (i < NPRED) ? sc[i] : 0.0f;
        sS[i] = (v > CONF_T) ? v : 0.0f;
        sI[i] = (i < NPRED) ? i : 0;
    }
#endif
    __syncthreads();
    for (unsigned k = 2; k <= NSORT; k <<= 1) {
        for (unsigned j = k >> 1; j > 0; j >>= 1) {
            for (int i = tid; i < NSORT; i += 1024) {
                unsigned l = (unsigned)i ^ j;
                if (l > (unsigned)i) {
                    float a_ = sS[i], c_ = sS[l];
                    bool desc = ((i & k) == 0);
                    bool sw = desc ? (a_ < c_) : (a_ > c_);
                    if (sw) {
                        sS[i] = c_; sS[l] = a_;
                        int t = sI[i]; sI[i] = sI[l]; sI[l] = t;
                    }
                }
            }
            __syncthreads();
        }
    }
    for (int i = tid; i < KPRE; i += 1024) {
        int idx = sI[i];
        cScores[(size_t)b * KPRE + i] = sS[i];
        cLabels[(size_t)b * KPRE + i] = labels[(size_t)b * NPRED + idx];
        size_t src = ((size_t)b * NPRED + idx) * 4;
        size_t dst = ((size_t)b * KPRE + i) * 4;
        cBoxes[dst + 0] = boxes[src + 0];
        cBoxes[dst + 1] = boxes[src + 1];
        cBoxes[dst + 2] = boxes[src + 2];
        cBoxes[dst + 3] = boxes[src + 3];
    }
}

// ---------------------------------------------------------------------------
// Greedy class-aware NMS (serial over i, parallel over j) + top-300 emit.
// One workgroup (256 thr) per image; candidates already score-descending.
// Candidate-box staging uses async load-to-LDS when available.
// ---------------------------------------------------------------------------
__global__ void __launch_bounds__(256)
yolo_nms(const float* __restrict__ cBoxes, const float* __restrict__ cScores,
         const int* __restrict__ cLabels, float* __restrict__ outBoxes,
         float* __restrict__ outScores, float* __restrict__ outLabels)
{
    __shared__ float sBB[KPRE * 4];
    __shared__ float sArea[KPRE];
    __shared__ float sSc[KPRE];
    __shared__ unsigned char sKeep[KPRE];
    const int b = blockIdx.x, tid = threadIdx.x;

#if USE_ASYNC_LDS
    // Async-DMA raw candidate boxes (16B per candidate) into LDS.
    for (int i = tid; i < KPRE; i += 256)
        __builtin_amdgcn_global_load_async_to_lds_b128(
            gbl_v4(cBoxes + ((size_t)b * KPRE + i) * 4),
            lds_v4(&sBB[i * 4]), 0, 0);
    __builtin_amdgcn_s_wait_asynccnt(0);
    __syncthreads();
    // In-place class-offset transform + area/keep init.
    for (int i = tid; i < KPRE; i += 256) {
        float off = (float)cLabels[(size_t)b * KPRE + i] * CLS_OFF;
        float x1 = sBB[i * 4 + 0] + off, y1 = sBB[i * 4 + 1] + off;
        float x2 = sBB[i * 4 + 2] + off, y2 = sBB[i * 4 + 3] + off;
        sBB[i * 4 + 0] = x1; sBB[i * 4 + 1] = y1;
        sBB[i * 4 + 2] = x2; sBB[i * 4 + 3] = y2;
        sArea[i] = fmaxf(x2 - x1, 0.0f) * fmaxf(y2 - y1, 0.0f);
        float scv = cScores[(size_t)b * KPRE + i];
        sSc[i] = scv;
        sKeep[i] = (scv > CONF_T) ? 1 : 0;
    }
#else
    for (int i = tid; i < KPRE; i += 256) {
        size_t s4 = ((size_t)b * KPRE + i) * 4;
        float off = (float)cLabels[(size_t)b * KPRE + i] * CLS_OFF;
        float x1 = cBoxes[s4 + 0] + off, y1 = cBoxes[s4 + 1] + off;
        float x2 = cBoxes[s4 + 2] + off, y2 = cBoxes[s4 + 3] + off;
        sBB[i * 4 + 0] = x1; sBB[i * 4 + 1] = y1;
        sBB[i * 4 + 2] = x2; sBB[i * 4 + 3] = y2;
        sArea[i] = fmaxf(x2 - x1, 0.0f) * fmaxf(y2 - y1, 0.0f);
        float scv = cScores[(size_t)b * KPRE + i];
        sSc[i] = scv;
        sKeep[i] = (scv > CONF_T) ? 1 : 0;
    }
#endif
    __syncthreads();
    for (int i = 0; i < KPRE; ++i) {
        if (sKeep[i]) {                            // uniform after barrier
            float bx1 = sBB[i * 4 + 0], by1 = sBB[i * 4 + 1];
            float bx2 = sBB[i * 4 + 2], by2 = sBB[i * 4 + 3];
            float ba  = sArea[i];
            for (int j = i + 1 + tid; j < KPRE; j += 256) {
                float ix1 = fmaxf(bx1, sBB[j * 4 + 0]);
                float iy1 = fmaxf(by1, sBB[j * 4 + 1]);
                float ix2 = fminf(bx2, sBB[j * 4 + 2]);
                float iy2 = fminf(by2, sBB[j * 4 + 3]);
                float inter = fmaxf(ix2 - ix1, 0.0f) * fmaxf(iy2 - iy1, 0.0f);
                float iou = inter / (ba + sArea[j] - inter + 1e-9f);
                if (iou > NMS_IOU) sKeep[j] = 0;
            }
        }
        __syncthreads();
    }
    if (tid == 0) {
        int cnt = 0;
        for (int i = 0; i < KPRE && cnt < MAXDET; ++i) {
            if (sKeep[i]) {
                size_t s4 = ((size_t)b * KPRE + i) * 4;
                size_t d4 = ((size_t)b * MAXDET + cnt) * 4;
                outBoxes[d4 + 0] = cBoxes[s4 + 0];
                outBoxes[d4 + 1] = cBoxes[s4 + 1];
                outBoxes[d4 + 2] = cBoxes[s4 + 2];
                outBoxes[d4 + 3] = cBoxes[s4 + 3];
                outScores[b * MAXDET + cnt] = sSc[i];
                outLabels[b * MAXDET + cnt] = (float)cLabels[(size_t)b * KPRE + i];
                ++cnt;
            }
        }
        for (; cnt < MAXDET; ++cnt) {
            size_t d4 = ((size_t)b * MAXDET + cnt) * 4;
            outBoxes[d4 + 0] = 0.0f; outBoxes[d4 + 1] = 0.0f;
            outBoxes[d4 + 2] = 0.0f; outBoxes[d4 + 3] = 0.0f;
            outScores[b * MAXDET + cnt] = 0.0f;
            outLabels[b * MAXDET + cnt] = -1.0f;
        }
    }
}

// ---------------------------------------------------------------------------
extern "C" void kernel_launch(void* const* d_in, const int* in_sizes, int n_in,
                              void* d_out, int out_size, void* d_ws, size_t ws_size,
                              hipStream_t stream)
{
    (void)in_sizes; (void)n_in; (void)out_size; (void)ws_size;
    const float* feat0 = (const float*)d_in[0];
    const float* w0    = (const float*)d_in[1];
    const float* b0    = (const float*)d_in[2];
    const float* feat1 = (const float*)d_in[3];
    const float* w1    = (const float*)d_in[4];
    const float* b1    = (const float*)d_in[5];
    const float* feat2 = (const float*)d_in[6];
    const float* w2    = (const float*)d_in[7];
    const float* b2    = (const float*)d_in[8];

    char* ws = (char*)d_ws;
    __bf16* wp0   = (__bf16*)(ws + 0);          //  64 KB (4 ktiles)
    __bf16* wp1   = (__bf16*)(ws + 65536);      // 128 KB (8 ktiles)
    __bf16* wp2   = (__bf16*)(ws + 196608);     // 256 KB (16 ktiles)
    float* boxes  = (float*)(ws + 458752);      // 16 x 25200 x 4 f32
    float* scores = (float*)(ws + 6909952);     // 16 x 25200 f32
    int*   labels = (int*)  (ws + 8522752);     // 16 x 25200 i32
    float* cBoxes = (float*)(ws + 10135552);    // 16 x 2048 x 4 f32
    float* cScores= (float*)(ws + 10659840);    // 16 x 2048 f32
    int*   cLabels= (int*)  (ws + 10790912);    // 16 x 2048 i32

    yolo_pack_w<<<(4  * 8192 + 255) / 256, 256, 0, stream>>>(w0, wp0, 4);
    yolo_pack_w<<<(8  * 8192 + 255) / 256, 256, 0, stream>>>(w1, wp1, 8);
    yolo_pack_w<<<(16 * 8192 + 255) / 256, 256, 0, stream>>>(w2, wp2, 16);

    yolo_decode<<<16 * 100, 128, 65536, stream>>>(feat0, wp0, b0, boxes, scores, labels,
        128, 6400, 80, 100, 8.0f, 0,     10.f, 13.f, 16.f, 30.f, 33.f, 23.f);
    yolo_decode<<<16 * 25,  128, 65536, stream>>>(feat1, wp1, b1, boxes, scores, labels,
        256, 1600, 40, 25,  16.0f, 19200, 30.f, 61.f, 62.f, 45.f, 59.f, 119.f);
    yolo_decode<<<16 * 7,   128, 65536, stream>>>(feat2, wp2, b2, boxes, scores, labels,
        512, 400,  20, 7,   32.0f, 24000, 116.f, 90.f, 156.f, 198.f, 373.f, 326.f);

    yolo_topk<<<16, 1024, NSORT * 8, stream>>>(scores, labels, boxes,
                                               cBoxes, cScores, cLabels);

    float* outBoxes  = (float*)d_out;
    float* outScores = outBoxes + 16 * MAXDET * 4;
    float* outLabels = outScores + 16 * MAXDET;
    yolo_nms<<<16, 256, 0, stream>>>(cBoxes, cScores, cLabels,
                                     outBoxes, outScores, outLabels);
}
